// TextAGCRN_27582279975075
// MI455X (gfx1250) — matrix operand order repaired and checked by reading, not compiled, not run on hardware
//
#include <hip/hip_runtime.h>
#include <stdint.h>

#define DEV static __device__ __forceinline__
typedef unsigned short u16;

typedef __attribute__((ext_vector_type(16))) __bf16 v16bf;
typedef __attribute__((ext_vector_type(8)))  float  v8f;

struct __attribute__((aligned(16))) U4 { unsigned x, y, z, w; };
union FragU { v16bf v; U4 q[2]; };

DEV u16 f2bf(float f) {
  union { float f; unsigned u; } x; x.f = f;
  unsigned r = x.u + 0x7fffu + ((x.u >> 16) & 1u);
  return (u16)(r >> 16);
}
DEV float bf2f(u16 h) {
  union { unsigned u; float f; } x; x.u = ((unsigned)h) << 16;
  return x.f;
}
DEV float gelu_f(float x)    { return 0.5f * x * (1.0f + erff(x * 0.70710678118654752f)); }
DEV float sigmoid_f(float x) { return 1.0f / (1.0f + __expf(-x)); }

// ---------------- epilogue modes ----------------
enum { EP_F32 = 0, EP_BF16 = 1, EP_XG = 2, EP_GATE = 3, EP_CAND = 4,
       EP_SCORES = 5, EP_VT = 6, EP_ATTNRES = 7 };
enum { ACT_NONE = 0, ACT_GELU = 1 };

struct GemmArgs {
  const u16* A;  long sA; int lda;
  const u16* B;  long sB; int ldb;   // B stored transposed: Bc[colN][K]
  void* C;       long sC; int ldc;
  const float* bias; long sBias;
  const float* x0;                   // extra read  (H for ATTNRES, R for CAND)
  float* x1;                         // extra write (R for GATE, H for CAND)
  int M, Nc, K, Nreal;
  float scale;
};

template<int EP, int ACT>
DEV void ta_store(const GemmArgs& ga, int z, const v8f& acc, int mb, int nb, int r, int g) {
  const int col = nb + r;
#pragma unroll
  for (int e = 0; e < 8; e++) {
    int row = mb + e + 8 * g;
    float val = acc[e];
    if (EP == EP_F32) {
      if (col < ga.Nreal) {
        if (ga.bias) val += ga.bias[(long)z * ga.sBias + col];
        if (ACT == ACT_GELU) val = gelu_f(val);
        ((float*)ga.C)[(long)z * ga.sC + (long)row * ga.ldc + col] = val;
      }
    } else if (EP == EP_BF16) {
      float o = 0.f;
      if (col < ga.Nreal) {
        if (ga.bias) val += ga.bias[(long)z * ga.sBias + col];
        if (ACT == ACT_GELU) val = gelu_f(val);
        o = val;
      }
      ((u16*)ga.C)[(long)z * ga.sC + (long)row * ga.ldc + col] = f2bf(o);
    } else if (EP == EP_XG) {           // row=node, col=b*66+c -> xg[node][b][66+c]
      int b = col / 66, c = col - b * 66;
      ((u16*)ga.C)[((long)row * 32 + b) * 160 + 66 + c] = f2bf(val);
    } else if (EP == EP_GATE) {         // z=node, row=batch; sigmoid; split Z|R
      val = sigmoid_f(val + ga.bias[(long)z * ga.sBias + col]);
      long off = (long)row * 65536 + (long)z * 64;
      if (col < 64) ((float*)ga.C)[off + col] = val;
      else          ga.x1[off + col - 64] = val;
    } else if (EP == EP_CAND) {         // tanh + GRU update, H in-place
      val = tanhf(val + ga.bias[(long)z * ga.sBias + col]);
      long off = (long)row * 65536 + (long)z * 64 + col;
      float Rv = ga.x0[off];
      float Hv = ga.x1[off];
      ga.x1[off] = Rv * Hv + (1.0f - Rv) * val;
    } else if (EP == EP_SCORES) {
      ((u16*)ga.C)[(long)z * ga.sC + (long)row * ga.ldc + col] = f2bf(val * ga.scale);
    } else if (EP == EP_VT) {           // row=b*1024+j, col=d -> vT[b][d][j]
      val += ga.bias[col];
      int b = row >> 10, j = row & 1023;
      ((u16*)ga.C)[(long)b * 65536 + (long)col * 1024 + j] = f2bf(val);
    } else if (EP == EP_ATTNRES) {      // += h (f32), write bf16
      val += ga.x0[(long)z * 65536 + (long)row * 64 + col];
      ((u16*)ga.C)[(long)z * ga.sC + (long)row * ga.ldc + col] = f2bf(val);
    }
  }
}

// TN GEMM: C[M,Nc] = A[M,K] @ Bc[Nc,K]^T ; bf16 in, f32 acc, one wave per 32x32 tile.
template<int EP, int ACT>
__global__ void ta_gemm(GemmArgs ga) {
  const int wpb  = blockDim.x >> 5;
  const int wave = threadIdx.x >> 5;
  const int lane = threadIdx.x & 31;
  const int tilesN = ga.Nc >> 5;
  const int tilesM = ga.M >> 5;
  int tile = blockIdx.x * wpb + wave;
  if (tile >= tilesM * tilesN) return;        // wave-uniform: EXEC stays all-ones
  const int z = blockIdx.z;
  const u16* Ab = ga.A + (long)z * ga.sA;
  const u16* Bb = ga.B + (long)z * ga.sB;
  const int tm = tile / tilesN;
  const int tn = tile - tm * tilesN;
  const int m0 = tm << 5, n0 = tn << 5;
  const int r = lane & 15, g = lane >> 4;
  v8f a00 = {0,0,0,0,0,0,0,0};
  v8f a01 = a00, a10 = a00, a11 = a00;
  const u16* pa0 = Ab + (long)(m0 + r) * ga.lda + 8 * g;   // A frag: K = 8g+0..7 | 16+8g+0..7
  const u16* pa1 = pa0 + (long)ga.lda * 16;
  const u16* pb0 = Bb + (long)(n0 + r) * ga.ldb + 8 * g;   // B frag: K = 8g+0..7 | 16+8g+0..7
  const u16* pb1 = pb0 + (long)ga.ldb * 16;
  for (int kb = 0; kb < ga.K; kb += 32) {
    FragU fa0, fa1, fb0, fb1;
    fa0.q[0] = *(const U4*)(pa0 + kb);
    fa0.q[1] = *(const U4*)(pa0 + kb + 16);
    fa1.q[0] = *(const U4*)(pa1 + kb);
    fa1.q[1] = *(const U4*)(pa1 + kb + 16);
    fb0.q[0] = *(const U4*)(pb0 + kb);
    fb0.q[1] = *(const U4*)(pb0 + kb + 16);
    fb1.q[0] = *(const U4*)(pb1 + kb);
    fb1.q[1] = *(const U4*)(pb1 + kb + 16);
    a00 = __builtin_amdgcn_wmma_f32_16x16x32_bf16(false, fa0.v, false, fb0.v, (short)0, a00, false, false);
    a01 = __builtin_amdgcn_wmma_f32_16x16x32_bf16(false, fa0.v, false, fb1.v, (short)0, a01, false, false);
    a10 = __builtin_amdgcn_wmma_f32_16x16x32_bf16(false, fa1.v, false, fb0.v, (short)0, a10, false, false);
    a11 = __builtin_amdgcn_wmma_f32_16x16x32_bf16(false, fa1.v, false, fb1.v, (short)0, a11, false, false);
  }
  ta_store<EP, ACT>(ga, z, a00, m0,      n0,      r, g);
  ta_store<EP, ACT>(ga, z, a01, m0,      n0 + 16, r, g);
  ta_store<EP, ACT>(ga, z, a10, m0 + 16, n0,      r, g);
  ta_store<EP, ACT>(ga, z, a11, m0 + 16, n0 + 16, r, g);
}

// ---------------- adjacency: A = row-softmax(relu(E E^T)), write bf16 ----------------
__global__ void ta_adj(const float* __restrict__ E, u16* __restrict__ Abf) {
  int n = blockIdx.x, tid = threadIdx.x;
  __shared__ float red[256];
  float en[10];
#pragma unroll
  for (int d = 0; d < 10; d++) en[d] = E[n * 10 + d];
  float vals[4]; float mx = -3.0e38f;
#pragma unroll
  for (int i = 0; i < 4; i++) {
    int m = tid + i * 256;
    float s = 0.f;
#pragma unroll
    for (int d = 0; d < 10; d++) s += en[d] * E[m * 10 + d];
    s = s > 0.f ? s : 0.f;
    vals[i] = s; mx = fmaxf(mx, s);
  }
  red[tid] = mx; __syncthreads();
  for (int o = 128; o > 0; o >>= 1) { if (tid < o) red[tid] = fmaxf(red[tid], red[tid + o]); __syncthreads(); }
  mx = red[0]; __syncthreads();
  float sum = 0.f;
#pragma unroll
  for (int i = 0; i < 4; i++) { vals[i] = __expf(vals[i] - mx); sum += vals[i]; }
  red[tid] = sum; __syncthreads();
  for (int o = 128; o > 0; o >>= 1) { if (tid < o) red[tid] += red[tid + o]; __syncthreads(); }
  float inv = 1.0f / red[0];
#pragma unroll
  for (int i = 0; i < 4; i++) Abf[(long)n * 1024 + tid + i * 256] = f2bf(vals[i] * inv);
}

// per-node weights: out[n][o][kk] = sum_d E[n,d]*Wp[d, kk/66, kk%66, o], pad kk>=132
__global__ void ta_node_w(const float* __restrict__ E, const float* __restrict__ Wp,
                          u16* __restrict__ out, int No, long total) {
  long idx = (long)blockIdx.x * blockDim.x + threadIdx.x;
  if (idx >= total) return;
  int kk = (int)(idx % 160);
  long rr = idx / 160;
  int o = (int)(rr % No);
  int n = (int)(rr / No);
  float acc = 0.f;
  if (kk < 132) {
    int k = kk / 66, i = kk - k * 66;
    const float* w = Wp + ((long)k * 66 + i) * No + o;
#pragma unroll
    for (int d = 0; d < 10; d++) acc += E[n * 10 + d] * w[(long)d * 2 * 66 * No];
    out[idx] = f2bf(acc);
  } else out[idx] = 0;
}

__global__ void ta_node_b(const float* __restrict__ E, const float* __restrict__ bp,
                          float* __restrict__ out, int No, long total) {
  long idx = (long)blockIdx.x * blockDim.x + threadIdx.x;
  if (idx >= total) return;
  int o = (int)(idx % No);
  int n = (int)(idx / No);
  float acc = 0.f;
#pragma unroll
  for (int d = 0; d < 10; d++) acc += E[n * 10 + d] * bp[d * No + o];
  out[idx] = acc;
}

// transpose+convert weight (Kreal x Nreal) -> Bc[Np][Kp] bf16 with zero pad
__global__ void ta_wT(const float* __restrict__ W, u16* __restrict__ Bc,
                      int Kreal, int Nreal, int Kp, int Np) {
  int idx = blockIdx.x * blockDim.x + threadIdx.x;
  if (idx >= Np * Kp) return;
  int k = idx % Kp, o = idx / Kp;
  float v = (k < Kreal && o < Nreal) ? W[k * Nreal + o] : 0.f;
  Bc[idx] = f2bf(v);
}

__global__ void ta_f2bf(const float* __restrict__ in, u16* __restrict__ out, long n) {
  long idx = (long)blockIdx.x * blockDim.x + threadIdx.x;
  if (idx < n) out[idx] = f2bf(in[idx]);
}

// pack Xc: XcT[j=b*66+c][m] and xg[m][b][c], c<2 from x[:,:,c,t], else H (optionally *Z)
__global__ void ta_pack(const float* __restrict__ x, const float* __restrict__ H,
                        const float* __restrict__ Z, u16* __restrict__ XcT,
                        u16* __restrict__ xg, int t, int cand) {
  long idx = (long)blockIdx.x * blockDim.x + threadIdx.x;
  if (idx >= 2112L * 1024) return;
  int m = (int)(idx & 1023);
  int j = (int)(idx >> 10);
  int b = j / 66, c = j - b * 66;
  long bm = (long)b * 1024 + m;
  float v;
  if (c < 2) v = x[(bm * 2 + c) * 12 + t];
  else {
    long ho = bm * 64 + (c - 2);
    v = H[ho];
    if (cand) v *= Z[ho];
  }
  u16 bv = f2bf(v);
  XcT[(long)j * 1024 + m] = bv;
  xg[((long)m * 32 + b) * 160 + c] = bv;
}

// layernorm over (N*G)=65536 per batch + gelu; write f32 and/or bf16
__global__ void ta_ln(const float* __restrict__ X, const float* __restrict__ gamma,
                      const float* __restrict__ beta, float* outF, u16* outB) {
  int b = blockIdx.x, tid = threadIdx.x;
  __shared__ float s1[256], s2[256];
  const float* xb = X + (long)b * 65536;
  float sum = 0.f, sq = 0.f;
  for (int i = tid; i < 65536; i += 256) { float v = xb[i]; sum += v; sq += v * v; }
  s1[tid] = sum; s2[tid] = sq; __syncthreads();
  for (int o = 128; o > 0; o >>= 1) {
    if (tid < o) { s1[tid] += s1[tid + o]; s2[tid] += s2[tid + o]; }
    __syncthreads();
  }
  float mu = s1[0] * (1.f / 65536.f);
  float var = s2[0] * (1.f / 65536.f) - mu * mu;
  float inv = rsqrtf(var + 1e-12f);
  for (int i = tid; i < 65536; i += 256) {
    float v = (xb[i] - mu) * inv * gamma[i] + beta[i];
    v = gelu_f(v);
    if (outF) outF[(long)b * 65536 + i] = v;
    if (outB) outB[(long)b * 65536 + i] = f2bf(v);
  }
}

// in-place bf16 row softmax, rows of 1024
__global__ void ta_softmax(u16* __restrict__ S) {
  long row = blockIdx.x;
  u16* s = S + row * 1024;
  int tid = threadIdx.x;
  __shared__ float red[256];
  float v[4]; float mx = -3.0e38f;
#pragma unroll
  for (int i = 0; i < 4; i++) { v[i] = bf2f(s[tid + i * 256]); mx = fmaxf(mx, v[i]); }
  red[tid] = mx; __syncthreads();
  for (int o = 128; o > 0; o >>= 1) { if (tid < o) red[tid] = fmaxf(red[tid], red[tid + o]); __syncthreads(); }
  mx = red[0]; __syncthreads();
  float sum = 0.f;
#pragma unroll
  for (int i = 0; i < 4; i++) { v[i] = __expf(v[i] - mx); sum += v[i]; }
  red[tid] = sum; __syncthreads();
  for (int o = 128; o > 0; o >>= 1) { if (tid < o) red[tid] += red[tid + o]; __syncthreads(); }
  float inv = 1.f / red[0];
#pragma unroll
  for (int i = 0; i < 4; i++) s[tid + i * 256] = f2bf(v[i] * inv);
}

// ---------------- host ----------------
static inline GemmArgs mkga(const u16* A, long sA, int lda, const u16* B, long sB, int ldb,
                            void* C, long sC, int ldc, const float* bias, long sBias,
                            int M, int Nc, int K, int Nreal, float scale,
                            const float* x0, float* x1) {
  GemmArgs g; g.A = A; g.sA = sA; g.lda = lda; g.B = B; g.sB = sB; g.ldb = ldb;
  g.C = C; g.sC = sC; g.ldc = ldc; g.bias = bias; g.sBias = sBias;
  g.x0 = x0; g.x1 = x1; g.M = M; g.Nc = Nc; g.K = K; g.Nreal = Nreal; g.scale = scale;
  return g;
}

template<int EP, int ACT>
static void run_gemm(hipStream_t s, const GemmArgs& ga, int zb, int threads) {
  int tiles = (ga.M >> 5) * (ga.Nc >> 5);
  int wpb = threads >> 5;
  dim3 grid((unsigned)((tiles + wpb - 1) / wpb), 1, (unsigned)zb);
  ta_gemm<EP, ACT><<<grid, dim3(threads), 0, s>>>(ga);
}

#define CDIV(a, b) (((a) + (b) - 1) / (b))

extern "C" void kernel_launch(void* const* d_in, const int* in_sizes, int n_in,
                              void* d_out, int out_size, void* d_ws, size_t ws_size,
                              hipStream_t stream) {
  (void)in_sizes; (void)n_in; (void)out_size; (void)ws_size;
  const float* x     = (const float*)d_in[0];
  const float* textp = (const float*)d_in[1];
  const float* E     = (const float*)d_in[2];
  const float* Wg    = (const float*)d_in[3];
  const float* bgp   = (const float*)d_in[4];
  const float* Wu    = (const float*)d_in[5];
  const float* bup   = (const float*)d_in[6];
  const float* ln1w  = (const float*)d_in[7];
  const float* ln1b  = (const float*)d_in[8];
  const float* dtW1  = (const float*)d_in[9];
  const float* dtb1  = (const float*)d_in[10];
  const float* dtW2  = (const float*)d_in[11];
  const float* dtb2  = (const float*)d_in[12];
  const float* dtW3  = (const float*)d_in[13];
  const float* dtb3  = (const float*)d_in[14];
  const float* tln1W = (const float*)d_in[15];
  const float* tln1b = (const float*)d_in[16];
  const float* tln2W = (const float*)d_in[17];
  const float* tln2b = (const float*)d_in[18];
  const float* ln2w  = (const float*)d_in[19];
  const float* ln2b  = (const float*)d_in[20];
  const float* kW    = (const float*)d_in[21];
  const float* kb    = (const float*)d_in[22];
  const float* vW    = (const float*)d_in[23];
  const float* vb    = (const float*)d_in[24];
  const float* alnw  = (const float*)d_in[25];
  const float* alnb  = (const float*)d_in[26];
  const float* lin1W = (const float*)d_in[27];
  const float* lin1b = (const float*)d_in[28];
  const float* lin2W = (const float*)d_in[29];
  const float* lin2b = (const float*)d_in[30];

  // ---- workspace carve (persistent) ----
  size_t off = 0;
  auto alloc = [&](size_t bytes) -> void* {
    off = (off + 255) & ~(size_t)255;
    void* p = (char*)d_ws + off; off += bytes; return p;
  };
  u16*   A_bf  = (u16*)  alloc(1024UL * 1024 * 2);
  u16*   WgT   = (u16*)  alloc(1024UL * 128 * 160 * 2);
  u16*   WuT   = (u16*)  alloc(1024UL * 64 * 160 * 2);
  float* bg    = (float*)alloc(1024UL * 128 * 4);
  float* bu    = (float*)alloc(1024UL * 64 * 4);
  float* H     = (float*)alloc(32UL * 1024 * 64 * 4);
  float* Zb    = (float*)alloc(32UL * 1024 * 64 * 4);
  float* Rb    = (float*)alloc(32UL * 1024 * 64 * 4);
  float* hcur  = (float*)alloc(32UL * 1024 * 64 * 4);
  u16*   h_bf  = (u16*)  alloc(32UL * 1024 * 64 * 2);
  u16*   t_bf  = (u16*)  alloc(32UL * 1024 * 64 * 2);
  u16*   hb2   = (u16*)  alloc(32UL * 1024 * 64 * 2);
  u16*   hb3   = (u16*)  alloc(32UL * 1024 * 64 * 2);
  u16*   W1c   = (u16*)  alloc(256UL * 768 * 2);
  u16*   W2c   = (u16*)  alloc(64UL * 256 * 2);
  u16*   W3c   = (u16*)  alloc(32UL * 64 * 2);
  u16*   tln1c = (u16*)  alloc(64UL * 32 * 2);
  u16*   kWc   = (u16*)  alloc(3UL * 64 * 64 * 2);
  u16*   vWc   = (u16*)  alloc(3UL * 64 * 64 * 2);
  u16*   tln2c = (u16*)  alloc(64UL * 64 * 2);
  u16*   lin1c = (u16*)  alloc(64UL * 64 * 2);
  u16*   lin2c = (u16*)  alloc(32UL * 64 * 2);

  // ---- phase-overlapped arena ----
  size_t arenaOff = (off + 255) & ~(size_t)255;
  auto arena = [&](size_t& o, size_t bytes) -> void* {
    o = (o + 255) & ~(size_t)255;
    void* p = (char*)d_ws + arenaOff + o; o += bytes; return p;
  };
  size_t o1 = 0;  // scan phase
  u16* XcT = (u16*)arena(o1, 2112UL * 1024 * 2);
  u16* xg  = (u16*)arena(o1, 1024UL * 32 * 160 * 2);
  size_t o2 = 0;  // text phase
  u16*   text_bf = (u16*)  arena(o2, 32768UL * 768 * 2);
  u16*   t1      = (u16*)  arena(o2, 32768UL * 256 * 2);
  u16*   t2      = (u16*)  arena(o2, 32768UL * 64 * 2);
  u16*   t3      = (u16*)  arena(o2, 32768UL * 32 * 2);
  float* tpre    = (float*)arena(o2, 32768UL * 64 * 4);
  size_t o3 = 0;  // attention phase
  u16*   scores = (u16*)  arena(o3, 32UL * 1024 * 1024 * 2);
  u16*   k_bf   = (u16*)  arena(o3, 32768UL * 64 * 2);
  u16*   vT     = (u16*)  arena(o3, 32768UL * 64 * 2);
  u16*   ha     = (u16*)  arena(o3, 32768UL * 64 * 2);
  float* hpre   = (float*)arena(o3, 32768UL * 64 * 4);

  // ---- prep (every launch; ws is poisoned between phases by the harness) ----
  hipMemsetAsync(H,  0, 32UL * 1024 * 64 * 4, stream);
  hipMemsetAsync(xg, 0, 1024UL * 32 * 160 * 2, stream);  // zero K-padding cols 132..159

  ta_adj<<<1024, 256, 0, stream>>>(E, A_bf);
  { long tot = 1024L * 128 * 160;
    ta_node_w<<<CDIV(tot, 256), 256, 0, stream>>>(E, Wg, WgT, 128, tot); }
  { long tot = 1024L * 64 * 160;
    ta_node_w<<<CDIV(tot, 256), 256, 0, stream>>>(E, Wu, WuT, 64, tot); }
  ta_node_b<<<CDIV(1024L * 128, 256), 256, 0, stream>>>(E, bgp, bg, 128, 1024L * 128);
  ta_node_b<<<CDIV(1024L * 64, 256), 256, 0, stream>>>(E, bup, bu, 64, 1024L * 64);
  ta_wT<<<CDIV(256 * 768, 256), 256, 0, stream>>>(dtW1, W1c, 768, 256, 768, 256);
  ta_wT<<<CDIV(64 * 256, 256), 256, 0, stream>>>(dtW2, W2c, 256, 64, 256, 64);
  ta_wT<<<CDIV(32 * 64, 256), 256, 0, stream>>>(dtW3, W3c, 64, 12, 64, 32);
  ta_wT<<<CDIV(64 * 32, 256), 256, 0, stream>>>(tln1W, tln1c, 12, 64, 32, 64);
  for (int i = 0; i < 3; i++) {
    ta_wT<<<CDIV(64 * 64, 256), 256, 0, stream>>>(kW + i * 4096, kWc + i * 4096, 64, 64, 64, 64);
    ta_wT<<<CDIV(64 * 64, 256), 256, 0, stream>>>(vW + i * 4096, vWc + i * 4096, 64, 64, 64, 64);
  }
  ta_wT<<<CDIV(64 * 64, 256), 256, 0, stream>>>(tln2W, tln2c, 64, 64, 64, 64);
  ta_wT<<<CDIV(64 * 64, 256), 256, 0, stream>>>(lin1W, lin1c, 64, 64, 64, 64);
  ta_wT<<<CDIV(32 * 64, 256), 256, 0, stream>>>(lin2W, lin2c, 64, 12, 64, 32);
  ta_f2bf<<<CDIV(32768L * 768, 256), 256, 0, stream>>>(textp, text_bf, 32768L * 768);

  // ---- AGCRN scan: 12 timesteps ----
  const long packBlocks = CDIV(2112L * 1024, 256);
  for (int t = 0; t < 12; t++) {
    // gate: Xc = [Xt, H]
    ta_pack<<<packBlocks, 256, 0, stream>>>(x, H, nullptr, XcT, xg, t, 0);
    { GemmArgs g = mkga(A_bf, 0, 1024, XcT, 0, 1024, xg, 0, 0, nullptr, 0,
                        1024, 2112, 1024, 2112, 1.f, nullptr, nullptr);
      run_gemm<EP_XG, ACT_NONE>(stream, g, 1, 256); }
    { GemmArgs g = mkga(xg, 32L * 160, 160, WgT, 128L * 160, 160, Zb, 0, 0,
                        bg, 128, 32, 128, 160, 128, 1.f, nullptr, Rb);
      run_gemm<EP_GATE, ACT_NONE>(stream, g, 1024, 128); }
    // candidate: Xc = [Xt, Z*H]; epilogue does H = R*H + (1-R)*tanh(...)
    ta_pack<<<packBlocks, 256, 0, stream>>>(x, H, Zb, XcT, xg, t, 1);
    { GemmArgs g = mkga(A_bf, 0, 1024, XcT, 0, 1024, xg, 0, 0, nullptr, 0,
                        1024, 2112, 1024, 2112, 1.f, nullptr, nullptr);
      run_gemm<EP_XG, ACT_NONE>(stream, g, 1, 256); }
    { GemmArgs g = mkga(xg, 32L * 160, 160, WuT, 64L * 160, 160, nullptr, 0, 0,
                        bu, 64, 32, 64, 160, 64, 1.f, Rb, H);
      run_gemm<EP_CAND, ACT_NONE>(stream, g, 1024, 64); }
  }
  // h = gelu(layernorm(H))
  ta_ln<<<32, 256, 0, stream>>>(H, ln1w, ln1b, hcur, h_bf);

  // ---- text MLP ----
  { GemmArgs g = mkga(text_bf, 0, 768, W1c, 0, 768, t1, 0, 256, dtb1, 0,
                      32768, 256, 768, 256, 1.f, nullptr, nullptr);
    run_gemm<EP_BF16, ACT_GELU>(stream, g, 1, 256); }
  { GemmArgs g = mkga(t1, 0, 256, W2c, 0, 256, t2, 0, 64, dtb2, 0,
                      32768, 64, 256, 64, 1.f, nullptr, nullptr);
    run_gemm<EP_BF16, ACT_GELU>(stream, g, 1, 256); }
  { GemmArgs g = mkga(t2, 0, 64, W3c, 0, 64, t3, 0, 32, dtb3, 0,
                      32768, 32, 64, 12, 1.f, nullptr, nullptr);
    run_gemm<EP_BF16, ACT_GELU>(stream, g, 1, 256); }
  { GemmArgs g = mkga(t3, 0, 32, tln1c, 0, 32, tpre, 0, 64, tln1b, 0,
                      32768, 64, 32, 64, 1.f, nullptr, nullptr);
    run_gemm<EP_F32, ACT_NONE>(stream, g, 1, 256); }
  ta_ln<<<32, 256, 0, stream>>>(tpre, ln2w, ln2b, nullptr, t_bf);

  // ---- attention layers ----
  for (int i = 0; i < 3; i++) {
    { GemmArgs g = mkga(h_bf, 0, 64, kWc + i * 4096, 0, 64, k_bf, 0, 64,
                        kb + i * 64, 0, 32768, 64, 64, 64, 1.f, nullptr, nullptr);
      run_gemm<EP_BF16, ACT_NONE>(stream, g, 1, 256); }
    { GemmArgs g = mkga(h_bf, 0, 64, vWc + i * 4096, 0, 64, vT, 0, 0,
                        vb + i * 64, 0, 32768, 64, 64, 64, 1.f, nullptr, nullptr);
      run_gemm<EP_VT, ACT_NONE>(stream, g, 1, 256); }
    { GemmArgs g = mkga(t_bf, 65536, 64, k_bf, 65536, 64, scores, 1048576, 1024,
                        nullptr, 0, 1024, 1024, 64, 1024, 0.125f, nullptr, nullptr);
      run_gemm<EP_SCORES, ACT_NONE>(stream, g, 32, 256); }
    ta_softmax<<<32768, 256, 0, stream>>>(scores);
    { GemmArgs g = mkga(scores, 1048576, 1024, vT, 65536, 1024, ha, 65536, 64,
                        nullptr, 0, 1024, 64, 1024, 64, 1.f, hcur, nullptr);
      run_gemm<EP_ATTNRES, ACT_NONE>(stream, g, 32, 256); }
    { GemmArgs g = mkga(ha, 0, 64, vWc + i * 4096, 0, 64, hpre, 0, 64,
                        vb + i * 64, 0, 32768, 64, 64, 64, 1.f, nullptr, nullptr);
      run_gemm<EP_F32, ACT_NONE>(stream, g, 1, 256); }
    ta_ln<<<32, 256, 0, stream>>>(hpre, alnw + i * 65536, alnb + i * 65536, hcur, h_bf);
  }

  // ---- head ----
  { GemmArgs g = mkga(h_bf, 0, 64, tln2c, 0, 64, hb2, 0, 64, tln2b, 0,
                      32768, 64, 64, 64, 1.f, nullptr, nullptr);
    run_gemm<EP_BF16, ACT_GELU>(stream, g, 1, 256); }
  { GemmArgs g = mkga(hb2, 0, 64, lin1c, 0, 64, hb3, 0, 64, lin1b, 0,
                      32768, 64, 64, 64, 1.f, nullptr, nullptr);
    run_gemm<EP_BF16, ACT_GELU>(stream, g, 1, 256); }
  { GemmArgs g = mkga(hb3, 0, 64, lin2c, 0, 64, d_out, 0, 12, lin2b, 0,
                      32768, 32, 64, 12, 1.f, nullptr, nullptr);
    run_gemm<EP_F32, ACT_NONE>(stream, g, 1, 256); }
}
